// MultiHeadSelfAttention_81458349736253
// MI455X (gfx1250) — compile-verified, matrix operand order checked by reading, not run on hardware
//
#include <hip/hip_runtime.h>

// Problem constants (match reference)
constexpr int Bsz = 2;
constexpr int Sq  = 2048;
constexpr int Dm  = 1024;
constexpr int NH  = 16;
constexpr int HSz = 64;   // head size; SCALE = 8 -> 1/SCALE = 0.125

typedef __attribute__((ext_vector_type(16))) __bf16 v16bf;
typedef __attribute__((ext_vector_type(8)))  float  v8f;

union FragA { v16bf v; unsigned int u[8]; };
union FragC { v8f  v; float f[8]; };

__device__ __forceinline__ unsigned short f2bf(float x) {
  unsigned int u = __float_as_uint(x);
  u += 0x7FFFu + ((u >> 16) & 1u);   // round-to-nearest-even
  return (unsigned short)(u >> 16);
}

// ISA 7.12.2 16-bit A/B fragment K-offset for uint (2 x bf16) slot vv, lane half hi.
// lanes 0-15: VGPR0-3 -> K0..7, VGPR4-7 -> K16..23 ; lanes 16-31: +8
__device__ __forceinline__ int kmap(int vv, int hi) {
  return ((vv & 3) << 1) + ((vv >> 2) << 4) + (hi << 3);
}

// LDS byte offset of a __shared__ object (wave-relative, for async-to-LDS VDST)
typedef __attribute__((address_space(3))) unsigned char lds_byte;
__device__ __forceinline__ unsigned lds_addr_of(const void* p) {
  return (unsigned)(unsigned long long)(lds_byte*)p;
}

// CDNA5 async copy: global -> LDS, 16B per lane, tracked by ASYNCcnt
__device__ __forceinline__ void async_b128(unsigned lds, const void* gptr) {
  asm volatile("global_load_async_to_lds_b128 %0, %1, off"
               :: "v"(lds), "v"((unsigned long long)gptr)
               : "memory");
}

// ---------------------------------------------------------------------------
// Kernel 1: QKV projection.  out[m,n] = x[m,:] . W[n,:] + bias[n]   (torch Linear)
// Block tile 128(M) x 128(N), K-step 64, 8 waves each computing 32x64.
// Q,K stored bf16 [B,H,S,HS]; V stored bf16 transposed [B,H,HS,S].
// (keeps VGPR staging: needs fp32 -> bf16 conversion in flight)
// ---------------------------------------------------------------------------
__global__ __launch_bounds__(256) void qkv_kernel(
    const float* __restrict__ x,
    const float* __restrict__ Wq, const float* __restrict__ bq,
    const float* __restrict__ Wk, const float* __restrict__ bk,
    const float* __restrict__ Wv, const float* __restrict__ bv,
    unsigned short* __restrict__ Qb, unsigned short* __restrict__ Kb,
    unsigned short* __restrict__ Vb) {
  const int z = blockIdx.z;
  const float* W  = (z == 0) ? Wq : (z == 1) ? Wk : Wv;
  const float* bs = (z == 0) ? bq : (z == 1) ? bk : bv;

  __shared__ __align__(16) unsigned short As[128][72];  // x tile [m][k]
  __shared__ __align__(16) unsigned short Bs[128][72];  // W tile [n][k]

  const int tid  = threadIdx.x;
  const int wave = tid >> 5, lane = tid & 31;
  const int hi   = lane >> 4, l16 = lane & 15;
  const int wm   = wave & 3,  wn  = wave >> 2;
  const int tile_m = blockIdx.y * 128;
  const int tile_n = blockIdx.x * 128;

  FragC acc[2][4];
#pragma unroll
  for (int i = 0; i < 2; ++i)
#pragma unroll
    for (int j = 0; j < 4; ++j)
#pragma unroll
      for (int r = 0; r < 8; ++r) acc[i][j].f[r] = 0.f;

  const int lr = tid >> 1;          // 128 rows, 2 threads per row
  const int lc = (tid & 1) * 32;    // each thread: 32 contiguous K elements

  for (int k0 = 0; k0 < Dm; k0 += 64) {
    __syncthreads();
#pragma unroll
    for (int j = 0; j < 32; ++j) {
      As[lr][lc + j] = f2bf(x[(size_t)(tile_m + lr) * Dm + k0 + lc + j]);
      Bs[lr][lc + j] = f2bf(W[(size_t)(tile_n + lr) * Dm + k0 + lc + j]);
    }
    __syncthreads();

#pragma unroll
    for (int f = 0; f < 2; ++f) {
      const int kk = f * 32;
      FragA a[2], b[4];
#pragma unroll
      for (int fm = 0; fm < 2; ++fm) {
        const int row = wm * 32 + fm * 16 + l16;
#pragma unroll
        for (int vv = 0; vv < 8; ++vv)
          a[fm].u[vv] = *(const unsigned int*)&As[row][kk + kmap(vv, hi)];
      }
#pragma unroll
      for (int fn = 0; fn < 4; ++fn) {
        const int row = wn * 64 + fn * 16 + l16;
#pragma unroll
        for (int vv = 0; vv < 8; ++vv)
          b[fn].u[vv] = *(const unsigned int*)&Bs[row][kk + kmap(vv, hi)];
      }
#pragma unroll
      for (int fm = 0; fm < 2; ++fm)
#pragma unroll
        for (int fn = 0; fn < 4; ++fn)
          acc[fm][fn].v = __builtin_amdgcn_wmma_f32_16x16x32_bf16(
              false, a[fm].v, false, b[fn].v, (short)0, acc[fm][fn].v, false, false);
    }
  }

  // Epilogue: add bias, round to bf16, scatter into head-major layouts.
#pragma unroll
  for (int fm = 0; fm < 2; ++fm)
#pragma unroll
    for (int fn = 0; fn < 4; ++fn)
#pragma unroll
      for (int r = 0; r < 8; ++r) {
        const int grow = tile_m + wm * 32 + fm * 16 + r + 8 * hi;  // token index
        const int gcol = tile_n + wn * 64 + fn * 16 + l16;          // feature index
        const float val = acc[fm][fn].f[r] + bs[gcol];
        const int bi = grow / Sq, si = grow % Sq;
        const int h  = gcol / HSz, d = gcol % HSz;
        const unsigned short e = f2bf(val);
        if (z == 0)
          Qb[(((size_t)bi * NH + h) * Sq + si) * HSz + d] = e;
        else if (z == 1)
          Kb[(((size_t)bi * NH + h) * Sq + si) * HSz + d] = e;
        else
          Vb[(((size_t)bi * NH + h) * HSz + d) * Sq + si] = e;  // V^T
      }
}

// ---------------------------------------------------------------------------
// Kernel 2: flash attention with online softmax.
// One workgroup (4 waves) per (b,h, 64-row q tile); wave w owns q rows w*16..+15.
// K / V^T tiles are double-buffered in LDS, filled by GLOBAL_LOAD_ASYNC_TO_LDS
// (ASYNCcnt), so the global->LDS fetch of tile kt+1 overlaps the WMMAs of kt.
// ---------------------------------------------------------------------------
__global__ __launch_bounds__(128) void attn_kernel(
    const unsigned short* __restrict__ Qb, const unsigned short* __restrict__ Kb,
    const unsigned short* __restrict__ Vb, float* __restrict__ out) {
  __shared__ __align__(16) unsigned short Ks[2][64][72];     // K tile  [key][hs]
  __shared__ __align__(16) unsigned short Vs[2][64][72];     // V^T tile [hs][key]
  __shared__ __align__(16) unsigned short Ps[4][16][72];     // per-wave P staging

  const int tid  = threadIdx.x;
  const int wave = tid >> 5, lane = tid & 31;
  const int hi   = lane >> 4, l16 = lane & 15;

  const int qt = blockIdx.x;
  const int bh = blockIdx.y;
  const int b  = bh / NH, h = bh % NH;
  const int qbase = qt * 64;

  const unsigned short* Qh = Qb + (size_t)bh * Sq * HSz;
  const unsigned short* Kh = Kb + (size_t)bh * Sq * HSz;
  const unsigned short* Vh = Vb + (size_t)bh * HSz * Sq;

  // Issue async global->LDS copies for one 64x64 bf16 K tile + V^T tile.
  // 512 16B-chunks per tile; 4 chunks per thread -> 8 async instrs per wave.
  auto issue_tile = [&](int kt2, int buf) {
    const unsigned short* Ksrc = Kh + (size_t)(kt2 * 64) * HSz;
    const unsigned short* Vsrc = Vh + kt2 * 64;
#pragma unroll
    for (int j = 0; j < 4; ++j) {
      const int chunk = tid * 4 + j;
      const int r = chunk >> 3, c = (chunk & 7) * 8;   // 8 ushorts = 16B
      async_b128(lds_addr_of(&Ks[buf][r][c]), Ksrc + (size_t)r * HSz + c);
      async_b128(lds_addr_of(&Vs[buf][r][c]), Vsrc + (size_t)r * Sq + c);
    }
  };

  // Q fragments in registers for this wave's 16 rows (K = HS = 64 -> 2 frags)
  FragA qa[2];
  {
    const int row = qbase + wave * 16 + l16;
#pragma unroll
    for (int f = 0; f < 2; ++f)
#pragma unroll
      for (int vv = 0; vv < 8; ++vv)
        qa[f].u[vv] =
            *(const unsigned int*)(Qh + (size_t)row * HSz + f * 32 + kmap(vv, hi));
  }

  float mrow[8], lrow[8];
  FragC o[4];
#pragma unroll
  for (int r = 0; r < 8; ++r) { mrow[r] = -1e30f; lrow[r] = 0.f; }
#pragma unroll
  for (int nt = 0; nt < 4; ++nt)
#pragma unroll
    for (int r = 0; r < 8; ++r) o[nt].f[r] = 0.f;

  issue_tile(0, 0);   // prologue prefetch

  for (int kt = 0; kt <= qt; ++kt) {
    const int buf = kt & 1;
    __syncthreads();  // all waves done reading buf^1 from iteration kt-1
    if (kt < qt) {
      issue_tile(kt + 1, buf ^ 1);                       // overlaps compute of kt
      asm volatile("s_wait_asynccnt 0x8" ::: "memory");  // tile kt complete
    } else {
      asm volatile("s_wait_asynccnt 0x0" ::: "memory");
    }
    __syncthreads();  // tile kt visible to every wave

    // S = (Q K^T) / sqrt(HS)
    FragC sc[4];
#pragma unroll
    for (int nt = 0; nt < 4; ++nt)
#pragma unroll
      for (int r = 0; r < 8; ++r) sc[nt].f[r] = 0.f;

#pragma unroll
    for (int f = 0; f < 2; ++f) {
      const int kk = f * 32;
#pragma unroll
      for (int nt = 0; nt < 4; ++nt) {
        FragA kb;
        const int n = nt * 16 + l16;  // key column of B = K row
#pragma unroll
        for (int vv = 0; vv < 8; ++vv)
          kb.u[vv] = *(const unsigned int*)&Ks[buf][n][kk + kmap(vv, hi)];
        sc[nt].v = __builtin_amdgcn_wmma_f32_16x16x32_bf16(
            false, qa[f].v, false, kb.v, (short)0, sc[nt].v, false, false);
      }
    }

    // scale + causal mask (only the diagonal tile needs masking)
#pragma unroll
    for (int nt = 0; nt < 4; ++nt)
#pragma unroll
      for (int r = 0; r < 8; ++r) {
        float sv = sc[nt].f[r] * 0.125f;
        if (kt == qt) {
          const int keyg = kt * 64 + nt * 16 + l16;
          const int qg   = qbase + wave * 16 + r + 8 * hi;
          if (keyg > qg) sv = -1e30f;
        }
        sc[nt].f[r] = sv;
      }

    // online softmax: per-row max / sum via 16-lane-half shuffles
#pragma unroll
    for (int r = 0; r < 8; ++r) {
      float mt = sc[0].f[r];
#pragma unroll
      for (int nt = 1; nt < 4; ++nt) mt = fmaxf(mt, sc[nt].f[r]);
#pragma unroll
      for (int off = 1; off < 16; off <<= 1) mt = fmaxf(mt, __shfl_xor(mt, off, 32));
      const float mn    = fmaxf(mrow[r], mt);
      const float alpha = __expf(mrow[r] - mn);
      mrow[r] = mn;
      float ps = 0.f;
#pragma unroll
      for (int nt = 0; nt < 4; ++nt) {
        const float p = __expf(sc[nt].f[r] - mn);
        sc[nt].f[r] = p;
        ps += p;
      }
#pragma unroll
      for (int off = 1; off < 16; off <<= 1) ps += __shfl_xor(ps, off, 32);
      lrow[r] = lrow[r] * alpha + ps;
#pragma unroll
      for (int nt = 0; nt < 4; ++nt) o[nt].f[r] *= alpha;
    }

    // C-layout -> A-layout for P via per-wave LDS staging (bf16)
#pragma unroll
    for (int nt = 0; nt < 4; ++nt)
#pragma unroll
      for (int r = 0; r < 8; ++r)
        Ps[wave][r + 8 * hi][nt * 16 + l16] = f2bf(sc[nt].f[r]);
    asm volatile("s_wait_dscnt 0" ::: "memory");

    // O += P V   (B operand columns = rows of V^T -> row-major LDS reads)
#pragma unroll
    for (int f = 0; f < 2; ++f) {
      const int kk = f * 32;
      FragA pa;
#pragma unroll
      for (int vv = 0; vv < 8; ++vv)
        pa.u[vv] = *(const unsigned int*)&Ps[wave][l16][kk + kmap(vv, hi)];
#pragma unroll
      for (int nt = 0; nt < 4; ++nt) {
        FragA vf;
        const int n = nt * 16 + l16;  // hs column of B = V^T row
#pragma unroll
        for (int vv = 0; vv < 8; ++vv)
          vf.u[vv] = *(const unsigned int*)&Vs[buf][n][kk + kmap(vv, hi)];
        o[nt].v = __builtin_amdgcn_wmma_f32_16x16x32_bf16(
            false, pa.v, false, vf.v, (short)0, o[nt].v, false, false);
      }
    }
  }

  // normalize and store fp32 output [B, S, H*HS]
#pragma unroll
  for (int nt = 0; nt < 4; ++nt)
#pragma unroll
    for (int r = 0; r < 8; ++r) {
      const int row = qbase + wave * 16 + r + 8 * hi;
      const int col = h * HSz + nt * 16 + l16;
      out[((size_t)b * Sq + row) * Dm + col] = o[nt].f[r] / lrow[r];
    }
}

// ---------------------------------------------------------------------------
extern "C" void kernel_launch(void* const* d_in, const int* in_sizes, int n_in,
                              void* d_out, int out_size, void* d_ws, size_t ws_size,
                              hipStream_t stream) {
  (void)in_sizes; (void)n_in; (void)out_size; (void)ws_size;
  const float* x  = (const float*)d_in[0];
  const float* Wq = (const float*)d_in[1];
  const float* bq = (const float*)d_in[2];
  const float* Wk = (const float*)d_in[3];
  const float* bk = (const float*)d_in[4];
  const float* Wv = (const float*)d_in[5];
  const float* bv = (const float*)d_in[6];
  float* out = (float*)d_out;

  const size_t qkv_elems = (size_t)Bsz * Sq * Dm;  // 4,194,304 bf16 each
  unsigned short* Qb = (unsigned short*)d_ws;
  unsigned short* Kb = Qb + qkv_elems;
  unsigned short* Vb = Kb + qkv_elems;             // 25.2 MB total workspace

  dim3 g1(Dm / 128, (Bsz * Sq) / 128, 3);          // 8 x 32 x 3
  qkv_kernel<<<g1, 256, 0, stream>>>(x, Wq, bq, Wk, bk, Wv, bv, Qb, Kb, Vb);

  dim3 g2(Sq / 64, Bsz * NH);                      // 32 x 32
  attn_kernel<<<g2, 128, 0, stream>>>(Qb, Kb, Vb, out);
}
